// FrequencyStream_71992241815981
// MI455X (gfx1250) — compile-verified, matrix operand order checked by reading, not run on hardware
//
#include <hip/hip_runtime.h>
#include <hip/hip_bf16.h>
#include <math.h>

// ---------------------------------------------------------------------------
// Types for CDNA5 WMMA (wave32): 16x16x32 bf16 -> f32
// ---------------------------------------------------------------------------
typedef __attribute__((ext_vector_type(16))) __bf16 v16bf;
typedef __attribute__((ext_vector_type(8)))  __bf16 bf16x8;
typedef __attribute__((ext_vector_type(8)))  float  v8f;
typedef __attribute__((ext_vector_type(4)))  float  f32x4;

__device__ __forceinline__ v8f wmma_bf16(v16bf a, v16bf b, v8f c) {
  // (neg_a, A, neg_b, B, c_mod, C, reuse_a, reuse_b)
  return __builtin_amdgcn_wmma_f32_16x16x32_bf16(false, a, false, b, (short)0, c, false, false);
}

// K offset inside a 32-wide K slice for VGPR pair v (0..7) and lane K-half kh.
__device__ __forceinline__ int kbase_of(int v, int kh) {
  return ((v < 4) ? 0 : 16) + kh * 8 + (v & 3) * 2;
}

// Per-lane fragment elements are K = kh*8+{0..7} (frag[0..7]) and
// K = 16+kh*8+{0..7} (frag[8..15]) -> two contiguous 16-byte loads.
__device__ __forceinline__ v16bf load_frag_k32(const __bf16* __restrict__ p, int kh) {
  bf16x8 lo = *(const bf16x8*)(p + kh * 8);
  bf16x8 hi = *(const bf16x8*)(p + 16 + kh * 8);
  v16bf r;
#pragma unroll
  for (int i = 0; i < 8; ++i) { r[i] = lo[i]; r[8 + i] = hi[i]; }
  return r;
}

// Same, but source is f32 (converted to bf16 in registers): 4x b128 loads.
__device__ __forceinline__ v16bf load_frag_k32_f32(const float* __restrict__ p, int kh) {
  f32x4 a0 = *(const f32x4*)(p + kh * 8);
  f32x4 a1 = *(const f32x4*)(p + kh * 8 + 4);
  f32x4 a2 = *(const f32x4*)(p + 16 + kh * 8);
  f32x4 a3 = *(const f32x4*)(p + 16 + kh * 8 + 4);
  v16bf r;
#pragma unroll
  for (int i = 0; i < 4; ++i) {
    r[i]      = (__bf16)a0[i];
    r[4 + i]  = (__bf16)a1[i];
    r[8 + i]  = (__bf16)a2[i];
    r[12 + i] = (__bf16)a3[i];
  }
  return r;
}

#define NDIM   224
#define IMGSZ  (224 * 224)      // 50176
#define NIMG   96               // 32 batches * 3 channels
#define BATCH  32
#define FCK    200704           // 256*28*28

// ---------------------------------------------------------------------------
// Build orthonormal DCT-II matrix D[k][m] in bf16 (row-major, 224x224)
// ---------------------------------------------------------------------------
__global__ void build_dct_bf16(__bf16* __restrict__ Dbf) {
  int i = blockIdx.x * blockDim.x + threadIdx.x;
  if (i >= IMGSZ) return;
  int k = i / NDIM, m = i % NDIM;
  float val = sqrtf(2.0f / 224.0f) *
              cosf(3.14159265358979323846f * (float)(2 * m + 1) * (float)k / 448.0f);
  if (k == 0) val *= 0.70710678118654752f;
  Dbf[i] = (__bf16)val;
}

// ---------------------------------------------------------------------------
// Repack OIHW conv weights into [Cout][Kpad] bf16 with k = khw*Cin + ci
// ---------------------------------------------------------------------------
__global__ void prep_weights(const float* __restrict__ w, __bf16* __restrict__ A,
                             int Cin, int Cout, int Kpad) {
  int K = Cin * 9;
  int total = Cout * Kpad;
  for (int i = blockIdx.x * blockDim.x + threadIdx.x; i < total;
       i += gridDim.x * blockDim.x) {
    int co = i / Kpad, k = i % Kpad;
    float val = 0.0f;
    if (k < K) {
      int khw = k / Cin, ci = k % Cin;
      val = w[(size_t)(co * Cin + ci) * 9 + khw];
    }
    A[i] = (__bf16)val;
  }
}

// ---------------------------------------------------------------------------
// DCT stage 1: T = D @ X  (A via b128 loads; B gathered from f32, K-strided)
// grid (14 h-tiles, 96 images), block 224 = 7 waves, each wave 2 n-tiles
// ---------------------------------------------------------------------------
__global__ void dct_rows(const float* __restrict__ x, const __bf16* __restrict__ Dbf,
                         __bf16* __restrict__ T) {
  const int img  = blockIdx.y;
  const int h0   = blockIdx.x * 16;
  const int wave = threadIdx.x >> 5;
  const int l    = threadIdx.x & 31;
  const int m16  = l & 15;
  const int khl  = l >> 4;
  const float*  X    = x + (size_t)img * IMGSZ;
  __bf16*       Tout = T + (size_t)img * IMGSZ;

  for (int nt = wave; nt < 14; nt += 7) {
    const int n0 = nt * 16;
    v8f acc = {};
    for (int k0 = 0; k0 < NDIM; k0 += 32) {
      v16bf a = load_frag_k32(Dbf + (size_t)(h0 + m16) * NDIM + k0, khl);
      v16bf b;
#pragma unroll
      for (int v = 0; v < 8; ++v) {
        int kb = kbase_of(v, khl);
        b[2 * v]     = (__bf16)X[(size_t)(k0 + kb) * NDIM + n0 + m16];
        b[2 * v + 1] = (__bf16)X[(size_t)(k0 + kb + 1) * NDIM + n0 + m16];
      }
      acc = wmma_bf16(a, b, acc);
    }
#pragma unroll
    for (int r = 0; r < 8; ++r)
      Tout[(size_t)(h0 + khl * 8 + r) * NDIM + n0 + m16] = (__bf16)acc[r];
  }
}

// ---------------------------------------------------------------------------
// DCT stage 2: FQ = T @ D^T, output stored NHWC-interleaved: FQ[b][h][w][c].
// A and B fragments both contiguous in K -> all b128 loads.
// ---------------------------------------------------------------------------
__global__ void dct_cols(const __bf16* __restrict__ T, const __bf16* __restrict__ Dbf,
                         __bf16* __restrict__ FQ) {
  const int img  = blockIdx.y;           // b*3 + c
  const int bimg = img / 3, chan = img % 3;
  const int h0   = blockIdx.x * 16;
  const int wave = threadIdx.x >> 5;
  const int l    = threadIdx.x & 31;
  const int m16  = l & 15;
  const int khl  = l >> 4;
  const __bf16* Tin = T + (size_t)img * IMGSZ;
  __bf16*       Fo  = FQ + (size_t)bimg * IMGSZ * 3 + chan;

  for (int wt = wave; wt < 14; wt += 7) {
    const int w0 = wt * 16;
    v8f acc = {};
    for (int k0 = 0; k0 < NDIM; k0 += 32) {
      v16bf a = load_frag_k32(Tin + (size_t)(h0 + m16) * NDIM + k0, khl);
      v16bf b = load_frag_k32(Dbf + (size_t)(w0 + m16) * NDIM + k0, khl);
      acc = wmma_bf16(a, b, acc);
    }
#pragma unroll
    for (int r = 0; r < 8; ++r)
      Fo[((size_t)(h0 + khl * 8 + r) * NDIM + w0 + m16) * 3] = (__bf16)acc[r];
  }
}

// ---------------------------------------------------------------------------
// Fused conv1 (Cin=3, K=27 pad 32) + bias + ReLU + 2x2 maxpool.
// Input FQ is NHWC [b][224][224][3]; output H1 is NHWC [b][112][112][64].
// Block = 128 threads (4 waves): 16 co x 2 y x 32 x -> 16 co x 16 x pooled.
// ---------------------------------------------------------------------------
__global__ void conv1_fused(const __bf16* __restrict__ FQ, const __bf16* __restrict__ A1,
                            const float* __restrict__ bias, __bf16* __restrict__ H1) {
  const int xsegs  = 7;
  const int cotile = blockIdx.x / xsegs;
  const int xseg   = blockIdx.x % xsegs;
  const int ypair  = blockIdx.y;
  const int bimg   = blockIdx.z;
  const int wave   = threadIdx.x >> 5;
  const int l      = threadIdx.x & 31;
  const int m16    = l & 15;
  const int khl    = l >> 4;
  const int dy     = wave >> 1;
  const int dx16   = (wave & 1) * 16;
  const int y      = ypair * 2 + dy;
  const int xpos   = xseg * 32 + dx16 + m16;
  const int co0    = cotile * 16;

  __shared__ float lds[2][16][33];

  const __bf16* in = FQ + (size_t)bimg * IMGSZ * 3;

  v16bf a = load_frag_k32(A1 + (size_t)(co0 + m16) * 32, khl);
  v16bf b;
#pragma unroll
  for (int v = 0; v < 8; ++v) {
#pragma unroll
    for (int t = 0; t < 2; ++t) {
      int k = kbase_of(v, khl) + t;
      float val = 0.0f;
      if (k < 27) {
        int khw = k / 3, ci = k % 3;
        int yy = y + khw / 3 - 1, xx = xpos + khw % 3 - 1;
        if (yy >= 0 && yy < NDIM && xx >= 0 && xx < NDIM)
          val = (float)in[((size_t)yy * NDIM + xx) * 3 + ci];
      }
      b[2 * v + t] = (__bf16)val;
    }
  }
  v8f acc = {};
  acc = wmma_bf16(a, b, acc);

#pragma unroll
  for (int r = 0; r < 8; ++r) {
    int col = 8 * khl + r;
    float vv = acc[r] + bias[co0 + col];
    lds[dy][col][dx16 + m16] = vv > 0.0f ? vv : 0.0f;
  }
  __syncthreads();

  // NHWC store: consecutive lanes write consecutive channels.
  for (int idx = threadIdx.x; idx < 256; idx += 128) {
    int col = idx & 15, xo = idx >> 4;
    float p = fmaxf(fmaxf(lds[0][col][2 * xo], lds[0][col][2 * xo + 1]),
                    fmaxf(lds[1][col][2 * xo], lds[1][col][2 * xo + 1]));
    int xout = xseg * 16 + xo;
    if (xout < 112)
      H1[(((size_t)bimg * 112 + ypair) * 112 + xout) * 64 + co0 + col] = (__bf16)p;
  }
}

// ---------------------------------------------------------------------------
// Generic fused conv (Cin multiple of 32) + bias + ReLU + 2x2 maxpool.
// Input NHWC [b][H][W][Cin]; K ordered khw*Cin+ci -> B fragments are two
// contiguous b128 loads per lane. Output NHWC or NCHW (flag).
// ---------------------------------------------------------------------------
__global__ void conv_fused(const __bf16* __restrict__ IN, const __bf16* __restrict__ A,
                           const float* __restrict__ bias, __bf16* __restrict__ OUT,
                           int Cin, int Cout, int H, int out_nhwc) {
  const int W      = H;
  const int Hout   = H / 2;
  const int Wout   = W / 2;
  const int xsegs  = (W + 31) / 32;
  const int cotile = blockIdx.x / xsegs;
  const int xseg   = blockIdx.x % xsegs;
  const int ypair  = blockIdx.y;
  const int bimg   = blockIdx.z;
  const int wave   = threadIdx.x >> 5;
  const int l      = threadIdx.x & 31;
  const int m16    = l & 15;
  const int khl    = l >> 4;
  const int dy     = wave >> 1;
  const int dx16   = (wave & 1) * 16;
  const int y      = ypair * 2 + dy;
  const int xpos   = xseg * 32 + dx16 + m16;
  const int co0    = cotile * 16;
  const int Kp     = Cin * 9;

  __shared__ float lds[2][16][33];

  const __bf16* in   = IN + (size_t)bimg * Cin * H * W;
  const __bf16* Arow = A + (size_t)(co0 + m16) * Kp;

  v8f acc = {};
  for (int khw = 0; khw < 9; ++khw) {
    const int yy  = y + khw / 3 - 1;
    const int xx  = xpos + khw % 3 - 1;
    const bool ib = (yy >= 0 && yy < H && xx >= 0 && xx < W);
    const __bf16* brow = in + ((size_t)yy * W + xx) * Cin;   // valid only if ib
    for (int cb = 0; cb < Cin; cb += 32) {
      v16bf a = load_frag_k32(Arow + khw * Cin + cb, khl);
      v16bf b;
      if (ib) b = load_frag_k32(brow + cb, khl);
      else    b = v16bf{};
      acc = wmma_bf16(a, b, acc);
    }
  }

#pragma unroll
  for (int r = 0; r < 8; ++r) {
    int col = 8 * khl + r;
    float vv = acc[r] + bias[co0 + col];
    lds[dy][col][dx16 + m16] = vv > 0.0f ? vv : 0.0f;
  }
  __syncthreads();

  for (int idx = threadIdx.x; idx < 256; idx += 128) {
    int col, xo;
    if (out_nhwc) { col = idx & 15; xo = idx >> 4; }
    else          { col = idx >> 4; xo = idx & 15; }
    float p = fmaxf(fmaxf(lds[0][col][2 * xo], lds[0][col][2 * xo + 1]),
                    fmaxf(lds[1][col][2 * xo], lds[1][col][2 * xo + 1]));
    int xout = xseg * 16 + xo;
    if (xout < Wout) {
      if (out_nhwc)
        OUT[(((size_t)bimg * Hout + ypair) * Wout + xout) * Cout + co0 + col] = (__bf16)p;
      else
        OUT[(((size_t)bimg * Cout + co0 + col) * Hout + ypair) * Wout + xout] = (__bf16)p;
    }
  }
}

// ---------------------------------------------------------------------------
// FC: out = fcb (init) + split-K bf16 WMMA with f32 atomic reduce.
// H3 is NCHW-flattened (matches reference reshape); fcw streamed once as f32.
// ---------------------------------------------------------------------------
__global__ void fc_init(float* __restrict__ out, const float* __restrict__ fcb) {
  int i = blockIdx.x * blockDim.x + threadIdx.x;
  if (i < BATCH * 512) out[i] = fcb[i & 511];
}

__global__ void fc_splitk(const __bf16* __restrict__ H3, const float* __restrict__ fcw,
                          float* __restrict__ out) {
  const int o0 = blockIdx.x * 16;
  const int m0 = blockIdx.z * 16;
  const int l  = threadIdx.x & 31;
  const int m16 = l & 15;
  const int khl = l >> 4;
  const int kstart = blockIdx.y * 2048;        // 98 chunks * 2048 = 200704

  const __bf16* Ar = H3  + (size_t)(m0 + m16) * FCK;
  const float*  Br = fcw + (size_t)(o0 + m16) * FCK;

  v8f acc = {};
  for (int k0 = kstart; k0 < kstart + 2048; k0 += 32) {
    __builtin_prefetch(Br + k0 + 512, 0, 1);   // stream fcw rows
    v16bf a = load_frag_k32(Ar + k0, khl);
    v16bf b = load_frag_k32_f32(Br + k0, khl);
    acc = wmma_bf16(a, b, acc);
  }
#pragma unroll
  for (int r = 0; r < 8; ++r) {
    int bb = m0 + 8 * khl + r;
    atomicAdd(&out[(size_t)bb * 512 + o0 + m16], acc[r]);
  }
}

// ---------------------------------------------------------------------------
// Host launcher
// ---------------------------------------------------------------------------
extern "C" void kernel_launch(void* const* d_in, const int* in_sizes, int n_in,
                              void* d_out, int out_size, void* d_ws, size_t ws_size,
                              hipStream_t stream) {
  const float* x   = (const float*)d_in[0];
  const float* w1  = (const float*)d_in[1];
  const float* b1  = (const float*)d_in[2];
  const float* w2  = (const float*)d_in[3];
  const float* b2  = (const float*)d_in[4];
  const float* w3  = (const float*)d_in[5];
  const float* b3  = (const float*)d_in[6];
  const float* fcw = (const float*)d_in[7];
  const float* fcb = (const float*)d_in[8];
  float* out = (float*)d_out;

  char* base = (char*)d_ws;
  size_t off = 0;
  auto carve = [&](size_t bytes) {
    char* p = base + off;
    off = (off + bytes + 255) & ~(size_t)255;
    return p;
  };
  __bf16* Dbf = (__bf16*)carve((size_t)IMGSZ * 2);
  __bf16* T   = (__bf16*)carve((size_t)NIMG * IMGSZ * 2);            // 9.6 MB
  __bf16* FQ  = (__bf16*)carve((size_t)NIMG * IMGSZ * 2);            // 9.6 MB (NHWC)
  __bf16* A1  = (__bf16*)carve((size_t)64 * 32 * 2);
  __bf16* A2  = (__bf16*)carve((size_t)128 * 576 * 2);
  __bf16* A3  = (__bf16*)carve((size_t)256 * 1152 * 2);
  __bf16* H1  = (__bf16*)carve((size_t)BATCH * 64 * 112 * 112 * 2);  // 51 MB (NHWC)
  __bf16* H2  = (__bf16*)carve((size_t)BATCH * 128 * 56 * 56 * 2);   // 26 MB (NHWC)
  __bf16* H3  = (__bf16*)carve((size_t)BATCH * 256 * 28 * 28 * 2);   // 13 MB (NCHW)
  if (off > ws_size) return;

  // 0) DCT basis + weight repack
  build_dct_bf16<<<(IMGSZ + 255) / 256, 256, 0, stream>>>(Dbf);
  prep_weights<<<64,   256, 0, stream>>>(w1, A1, 3,   64,  32);
  prep_weights<<<288,  256, 0, stream>>>(w2, A2, 64,  128, 576);
  prep_weights<<<1152, 256, 0, stream>>>(w3, A3, 128, 256, 1152);

  // 1) 2D DCT: T = D @ X ; FQ = T @ D^T (NHWC)
  dct_rows<<<dim3(14, NIMG), 224, 0, stream>>>(x, Dbf, T);
  dct_cols<<<dim3(14, NIMG), 224, 0, stream>>>(T, Dbf, FQ);

  // 2) conv1 + relu + pool : NHWC [32,112,112,64]
  conv1_fused<<<dim3(4 * 7, 112, BATCH), 128, 0, stream>>>(FQ, A1, b1, H1);

  // 3) conv2 + relu + pool : NHWC [32,56,56,128]
  conv_fused<<<dim3(8 * 4, 56, BATCH), 128, 0, stream>>>(H1, A2, b2, H2, 64, 128, 112, 1);

  // 4) conv3 + relu + pool : NCHW [32,256,28,28] (matches FC flatten order)
  conv_fused<<<dim3(16 * 2, 28, BATCH), 128, 0, stream>>>(H2, A3, b3, H3, 128, 256, 56, 0);

  // 5) FC: out = h3 @ fcw^T + fcb (split-K, atomic f32 reduce)
  fc_init<<<(BATCH * 512 + 255) / 256, 256, 0, stream>>>(out, fcb);
  fc_splitk<<<dim3(512 / 16, 98, BATCH / 16), 32, 0, stream>>>(H3, fcw, out);
}
